// Seq2Seq_16045997818378
// MI455X (gfx1250) — matrix operand order repaired and checked by reading, not compiled, not run on hardware
//
#include <hip/hip_runtime.h>
#include <hip/hip_bf16.h>
#include <cstdint>

// ---------------------------------------------------------------------------
// Seq2Seq (2-layer LSTM enc, bugged-decoder = batched single step, attention)
// for MI455X / gfx1250: bf16 WMMA (v_wmma_f32_16x16x32_bf16), fp32 accumulate.
// B=32, T_IN=400, T_OUT=16, V=32000, E=512, H=1024.
// Encoder recurrent weights precast to bf16 (L2-resident, reused 400x).
// ---------------------------------------------------------------------------

typedef __bf16 bf16_t;
typedef __attribute__((ext_vector_type(16))) __bf16 v16bf;
typedef __attribute__((ext_vector_type(8)))  __bf16 v8bf;
typedef __attribute__((ext_vector_type(8)))  float  v8f;

#define DEV static __device__ __forceinline__

DEV v8f wmma_bf16(v16bf a, v16bf b, v8f c) {
  // 8 args: (neg_a, A, neg_b, B, c_mod, C, reuse_a, reuse_b)
  return __builtin_amdgcn_wmma_f32_16x16x32_bf16(false, a, false, b, (short)0, c,
                                                 false, false);
}

DEV v16bf cat8(v8bf lo, v8bf hi) {
  return __builtin_shufflevector(lo, hi, 0, 1, 2, 3, 4, 5, 6, 7, 8, 9, 10, 11,
                                 12, 13, 14, 15);
}

DEV float sigf(float x) { return 1.0f / (1.0f + __expf(-x)); }

// A fragment: row-major [*, lda] bf16, tile rows m0..m0+15, cols k0..k0+31.
// Layout: lanes 0-15 (h=0): K = {0..15}; lanes 16-31: K = {16..31}.
DEV v16bf load_a(const bf16_t* A, long long lda, int m0, int k0, int lane) {
  int m = lane & 15, h = lane >> 4;
  const bf16_t* p = A + (long long)(m0 + m) * lda + k0 + h * 16;
  v8bf lo = *(const v8bf*)p;
  v8bf hi = *(const v8bf*)(p + 8);
  return cat8(lo, hi);
}

// B fragment (B^T pattern): W is [N_total, K_total] row-major; Btile[k][n] =
// W[n0+n][k0+k]. Lane n=L&15, element j -> k=h*16+j.
DEV v16bf load_bt_f32(const float* W, long long ldb, int n0, int k0, int lane) {
  int n = lane & 15, h = lane >> 4;
  const float* p = W + (long long)(n0 + n) * ldb + k0 + h * 16;
  const float4* q = (const float4*)p;
  float4 a0 = q[0], a1 = q[1], a2 = q[2], a3 = q[3];
  v16bf b;
  b[0] = (bf16_t)a0.x;  b[1] = (bf16_t)a0.y;  b[2] = (bf16_t)a0.z;  b[3] = (bf16_t)a0.w;
  b[4] = (bf16_t)a1.x;  b[5] = (bf16_t)a1.y;  b[6] = (bf16_t)a1.z;  b[7] = (bf16_t)a1.w;
  b[8] = (bf16_t)a2.x;  b[9] = (bf16_t)a2.y;  b[10] = (bf16_t)a2.z; b[11] = (bf16_t)a2.w;
  b[12] = (bf16_t)a3.x; b[13] = (bf16_t)a3.y; b[14] = (bf16_t)a3.z; b[15] = (bf16_t)a3.w;
  return b;
}

DEV v16bf load_bt_bf16(const bf16_t* W, long long ldb, int n0, int k0, int lane) {
  int n = lane & 15, h = lane >> 4;
  const bf16_t* p = W + (long long)(n0 + n) * ldb + k0 + h * 16;
  v8bf lo = *(const v8bf*)p;
  v8bf hi = *(const v8bf*)(p + 8);
  return cat8(lo, hi);
}

// B fragment (B is [K,N] bf16, not transposed): Btile[k][n] = B[k0+k][n0+n].
DEV v16bf load_bn_bf16(const bf16_t* B, long long ldb, int n0, int k0, int lane) {
  int n = lane & 15, h = lane >> 4;
  v16bf b;
#pragma unroll
  for (int j = 0; j < 16; ++j)
    b[j] = B[(long long)(k0 + h * 16 + j) * ldb + n0 + n];
  return b;
}

// ---------------------------------------------------------------------------
// Generic batched GEMM: C = act(A @ op(B) + bias1 + bias2 + addend)
// BMODE: 0 = B^T from fp32 [N,K]; 1 = B^T from bf16 [N,K]; 2 = B from bf16 [K,N]
// TILES: N-tiles per wave (A fragment reused TILES times). Block = 4 waves.
// TILES=4 requires N % 256 == 0 at the grid level.
// ---------------------------------------------------------------------------
template <int BMODE, int TILES>
__global__ void gemm_wmma(const bf16_t* __restrict__ A, long long lda, long long aB,
                          const void* Bp, long long ldb, long long bB,
                          float* __restrict__ Cf, bf16_t* __restrict__ Cb,
                          long long ldc, long long cB, int N, int K,
                          const float* __restrict__ bias1,
                          const float* __restrict__ bias2,
                          const float* __restrict__ addend, long long addLd,
                          int act) {
  int lane = threadIdx.x & 31;
  int wid = threadIdx.x >> 5;
  int n0 = (blockIdx.x * 4 + wid) * (16 * TILES);
  if (n0 >= N) return;
  int m0 = blockIdx.y * 16;
  int z = blockIdx.z;

  A += (long long)z * aB;
  const float* Bf = (const float*)Bp + (long long)z * bB;
  const bf16_t* Bh = (const bf16_t*)Bp + (long long)z * bB;

  v8f acc[TILES];
#pragma unroll
  for (int i = 0; i < TILES; ++i)
    acc[i] = (v8f){0.f, 0.f, 0.f, 0.f, 0.f, 0.f, 0.f, 0.f};

  for (int k0 = 0; k0 < K; k0 += 32) {
    v16bf a = load_a(A, lda, m0, k0, lane);
#pragma unroll
    for (int i = 0; i < TILES; ++i) {
      v16bf b;
      if (BMODE == 0)      b = load_bt_f32(Bf, ldb, n0 + i * 16, k0, lane);
      else if (BMODE == 1) b = load_bt_bf16(Bh, ldb, n0 + i * 16, k0, lane);
      else                 b = load_bn_bf16(Bh, ldb, n0 + i * 16, k0, lane);
      acc[i] = wmma_bf16(a, b, acc[i]);
    }
  }

  int half = lane >> 4;
  float* cf = Cf ? Cf + (long long)z * cB : nullptr;
  bf16_t* cb = Cb ? Cb + (long long)z * cB : nullptr;
#pragma unroll
  for (int i = 0; i < TILES; ++i) {
    int col = n0 + i * 16 + (lane & 15);
    float badd = 0.f;
    if (bias1) badd += bias1[col];
    if (bias2) badd += bias2[col];
#pragma unroll
    for (int r = 0; r < 8; ++r) {
      int row = m0 + r + half * 8;
      float v = acc[i][r] + badd;
      if (addend) v += addend[(long long)row * addLd + col];
      if (act == 1) v = tanhf(v);
      long long idx = (long long)row * ldc + col;
      if (cf) cf[idx] = v;
      if (cb) cb[idx] = (bf16_t)v;
    }
  }
}

// ---------------------------------------------------------------------------
// Fused encoder LSTM step (M=32): gates = hprev@Whh^T + xt@Wih^T + biases;
// elementwise LSTM; c updated in place; h written bf16 into history slot.
// Weights pre-cast bf16 (L2-resident). KX compile-time for a clean K loop.
// Block = 64 threads (2 waves, one per 16-row tile); grid.x = H/16 = 64.
// ---------------------------------------------------------------------------
template <int KX>
__global__ void enc_lstm_step(const bf16_t* __restrict__ xt,
                              const bf16_t* __restrict__ hprev,
                              const bf16_t* __restrict__ Wih,
                              const bf16_t* __restrict__ Whh,
                              const float* __restrict__ bih,
                              const float* __restrict__ bhh,
                              float* __restrict__ c, bf16_t* __restrict__ hout) {
  int lane = threadIdx.x & 31;
  int wid = threadIdx.x >> 5;
  int m0 = wid * 16;
  int nc = blockIdx.x * 16;

  v8f acc[4];
#pragma unroll
  for (int g = 0; g < 4; ++g) acc[g] = (v8f){0.f, 0.f, 0.f, 0.f, 0.f, 0.f, 0.f, 0.f};

#pragma unroll 2
  for (int k0 = 0; k0 < 1024; k0 += 32) {
    v16bf a = load_a(hprev, 1024, m0, k0, lane);
    // prefetch next K-slab of the recurrent weight row for this lane
    if (k0 + 32 < 1024)
      __builtin_prefetch(Whh + (long long)(nc + (lane & 15)) * 1024 + k0 + 32, 0, 3);
#pragma unroll
    for (int g = 0; g < 4; ++g) {
      v16bf b = load_bt_bf16(Whh, 1024, (g << 10) + nc, k0, lane);
      acc[g] = wmma_bf16(a, b, acc[g]);
    }
  }
#pragma unroll 2
  for (int k0 = 0; k0 < KX; k0 += 32) {
    v16bf a = load_a(xt, KX, m0, k0, lane);
#pragma unroll
    for (int g = 0; g < 4; ++g) {
      v16bf b = load_bt_bf16(Wih, KX, (g << 10) + nc, k0, lane);
      acc[g] = wmma_bf16(a, b, acc[g]);
    }
  }

  int col = nc + (lane & 15);
  int half = lane >> 4;
  float bi = bih[col] + bhh[col];
  float bf_ = bih[1024 + col] + bhh[1024 + col];
  float bg = bih[2048 + col] + bhh[2048 + col];
  float bo = bih[3072 + col] + bhh[3072 + col];
#pragma unroll
  for (int r = 0; r < 8; ++r) {
    int row = m0 + r + half * 8;
    float gi = sigf(acc[0][r] + bi);
    float gf = sigf(acc[1][r] + bf_);
    float gg = tanhf(acc[2][r] + bg);
    float go = sigf(acc[3][r] + bo);
    long long ci = (long long)row * 1024 + col;
    float cn = gf * c[ci] + gi * gg;
    c[ci] = cn;
    hout[ci] = (bf16_t)(go * tanhf(cn));
  }
}

// ---------------------------------------------------------------------------
// Batched decoder LSTM "step" (M=512 = B*T_OUT, all steps independent):
// gates = x@Wih^T + Hg[b]; c from cT[b] (broadcast per batch row group).
// Runs once per layer -> fp32 weights converted in flight.
// Block = 128 (4 waves over 4 row tiles); grid = (64 col tiles, 8 row groups)
// ---------------------------------------------------------------------------
template <int KX>
__global__ void dec_lstm_step(const bf16_t* __restrict__ x,
                              const float* __restrict__ Wih,
                              const float* __restrict__ Hg,  // [32,4096] incl. biases
                              const float* __restrict__ cT,  // [32,1024]
                              bf16_t* __restrict__ hout) {
  int lane = threadIdx.x & 31;
  int wid = threadIdx.x >> 5;
  int m0 = (blockIdx.y * 4 + wid) * 16;
  int nc = blockIdx.x * 16;

  v8f acc[4];
#pragma unroll
  for (int g = 0; g < 4; ++g) acc[g] = (v8f){0.f, 0.f, 0.f, 0.f, 0.f, 0.f, 0.f, 0.f};

  for (int k0 = 0; k0 < KX; k0 += 32) {
    v16bf a = load_a(x, KX, m0, k0, lane);
#pragma unroll
    for (int g = 0; g < 4; ++g) {
      v16bf b = load_bt_f32(Wih, KX, (g << 10) + nc, k0, lane);
      acc[g] = wmma_bf16(a, b, acc[g]);
    }
  }

  int col = nc + (lane & 15);
  int half = lane >> 4;
#pragma unroll
  for (int r = 0; r < 8; ++r) {
    int row = m0 + r + half * 8;
    int b = row >> 4;  // row = b*16 + t
    const float* hg = Hg + (long long)b * 4096;
    float gi = sigf(acc[0][r] + hg[col]);
    float gf = sigf(acc[1][r] + hg[1024 + col]);
    float gg = tanhf(acc[2][r] + hg[2048 + col]);
    float go = sigf(acc[3][r] + hg[3072 + col]);
    float cn = gf * cT[(long long)b * 1024 + col] + gi * gg;
    hout[(long long)row * 1024 + col] = (bf16_t)(go * tanhf(cn));
  }
}

// ---------------------------------------------------------------------------
// fp32 -> bf16 precast (encoder recurrent weights, done once per launch)
// ---------------------------------------------------------------------------
__global__ void cast_bf16_k(const float* __restrict__ in, bf16_t* __restrict__ out,
                            long long n) {
  long long i = ((long long)blockIdx.x * blockDim.x + threadIdx.x) * 4;
  if (i + 3 >= n) {
    for (long long j = i; j < n; ++j) out[j] = (bf16_t)in[j];
    return;
  }
  float4 v = *(const float4*)(in + i);
  v8bf dummy;
  (void)dummy;
  out[i + 0] = (bf16_t)v.x;
  out[i + 1] = (bf16_t)v.y;
  out[i + 2] = (bf16_t)v.z;
  out[i + 3] = (bf16_t)v.w;
}

// ---------------------------------------------------------------------------
// Softmax over 400 valid cols of a [512,416] fp32 matrix -> bf16 alpha,
// tail cols 400..415 zeroed. One wave per row.
// ---------------------------------------------------------------------------
__global__ void softmax400(const float* __restrict__ s, bf16_t* __restrict__ a) {
  int row = blockIdx.x;
  int lane = threadIdx.x;
  const float* sr = s + (long long)row * 416;
  bf16_t* ar = a + (long long)row * 416;

  float vals[13];
  int n = 0;
  float mx = -1e30f;
  for (int j = lane; j < 400; j += 32) {
    float v = sr[j];
    vals[n++] = v;
    mx = fmaxf(mx, v);
  }
#pragma unroll
  for (int off = 16; off; off >>= 1) mx = fmaxf(mx, __shfl_xor(mx, off, 32));
  float sum = 0.f;
  n = 0;
  for (int j = lane; j < 400; j += 32) {
    float e = __expf(vals[n] - mx);
    vals[n++] = e;
    sum += e;
  }
#pragma unroll
  for (int off = 16; off; off >>= 1) sum += __shfl_xor(sum, off, 32);
  float inv = 1.f / sum;
  n = 0;
  for (int j = lane; j < 400; j += 32) ar[j] = (bf16_t)(vals[n++] * inv);
  for (int j = 400 + lane; j < 416; j += 32) ar[j] = (bf16_t)0.f;
}

// ---------------------------------------------------------------------------
// Embedding gathers (fp32 table -> bf16 activations)
// ---------------------------------------------------------------------------
__global__ void embed_enc_k(const int* __restrict__ tok, const float* __restrict__ emb,
                            bf16_t* __restrict__ out) {
  long long i = (long long)blockIdx.x * blockDim.x + threadIdx.x;
  if (i >= 400LL * 32 * 512) return;
  int e = (int)(i & 511);
  long long r = i >> 9;
  int b = (int)(r & 31);
  int t = (int)(r >> 5);
  int tk = tok[b * 400 + t];
  out[i] = (bf16_t)emb[(long long)tk * 512 + e];  // layout [t][b][e]
}

__global__ void embed_dec_k(const int* __restrict__ titles, const float* __restrict__ emb,
                            bf16_t* __restrict__ out) {
  int i = blockIdx.x * blockDim.x + threadIdx.x;
  if (i >= 512 * 512) return;
  int e = i & 511;
  int row = i >> 9;  // b*16 + t
  int b = row >> 4;
  int t = row & 15;
  int tk = (t == 0) ? 2 : titles[b * 16 + t];
  out[i] = (bf16_t)emb[(long long)tk * 512 + e];
}

// ---------------------------------------------------------------------------
// Host driver
// ---------------------------------------------------------------------------
extern "C" void kernel_launch(void* const* d_in, const int* in_sizes, int n_in,
                              void* d_out, int out_size, void* d_ws, size_t ws_size,
                              hipStream_t stream) {
  const int* contents = (const int*)d_in[0];
  const int* titles   = (const int*)d_in[1];
  const float* emb_enc = (const float*)d_in[2];
  const float* emb_dec = (const float*)d_in[3];
  const float* eWih0 = (const float*)d_in[4];
  const float* eWhh0 = (const float*)d_in[5];
  const float* ebih0 = (const float*)d_in[6];
  const float* ebhh0 = (const float*)d_in[7];
  const float* eWih1 = (const float*)d_in[8];
  const float* eWhh1 = (const float*)d_in[9];
  const float* ebih1 = (const float*)d_in[10];
  const float* ebhh1 = (const float*)d_in[11];
  const float* dWih0 = (const float*)d_in[12];
  const float* dWhh0 = (const float*)d_in[13];
  const float* dbih0 = (const float*)d_in[14];
  const float* dbhh0 = (const float*)d_in[15];
  const float* dWih1 = (const float*)d_in[16];
  const float* dWhh1 = (const float*)d_in[17];
  const float* dbih1 = (const float*)d_in[18];
  const float* dbhh1 = (const float*)d_in[19];
  const float* w_in   = (const float*)d_in[20];
  const float* w_outW = (const float*)d_in[21];
  const float* w_outb = (const float*)d_in[22];
  const float* outW   = (const float*)d_in[23];
  const float* outb   = (const float*)d_in[24];

  // Workspace carve-up (256B aligned)
  uint8_t* base = (uint8_t*)d_ws;
  size_t off = 0;
  auto alloc = [&](size_t bytes) -> uint8_t* {
    off = (off + 255) & ~(size_t)255;
    uint8_t* p = base + off;
    off += bytes;
    return p;
  };
  const long long HB = 32 * 1024;  // per-time h stride (elements)
  bf16_t* enc_x    = (bf16_t*)alloc(400LL * 32 * 512 * 2);   // [t][b][e]
  bf16_t* zero_h   = (bf16_t*)alloc(HB * 2);
  float*  c0       = (float*)alloc(HB * 4);
  float*  c1       = (float*)alloc(HB * 4);
  bf16_t* h0_hist  = (bf16_t*)alloc(400LL * HB * 2);         // layer0 h history
  bf16_t* enc_hist = (bf16_t*)alloc(416LL * HB * 2);         // layer1 h history (+pad)
  bf16_t* dec_x    = (bf16_t*)alloc(512LL * 512 * 2);
  float*  Hg0      = (float*)alloc(32LL * 4096 * 4);
  float*  Hg1      = (float*)alloc(32LL * 4096 * 4);
  bf16_t* dec_h0   = (bf16_t*)alloc(512LL * 1024 * 2);
  bf16_t* dec_o    = (bf16_t*)alloc(512LL * 1024 * 2);
  bf16_t* energy   = (bf16_t*)alloc(512LL * 1024 * 2);
  float*  scores   = (float*)alloc(512LL * 416 * 4);
  bf16_t* alpha    = (bf16_t*)alloc(512LL * 416 * 2);
  bf16_t* context  = (bf16_t*)alloc(512LL * 1024 * 2);
  float*  tmp1     = (float*)alloc(512LL * 1024 * 4);
  bf16_t* attn_h   = (bf16_t*)alloc(512LL * 1024 * 2);
  // bf16 precast encoder weights (L2-resident across the 800 step kernels)
  bf16_t* bWih0 = (bf16_t*)alloc(4096LL * 512 * 2);
  bf16_t* bWhh0 = (bf16_t*)alloc(4096LL * 1024 * 2);
  bf16_t* bWih1 = (bf16_t*)alloc(4096LL * 1024 * 2);
  bf16_t* bWhh1 = (bf16_t*)alloc(4096LL * 1024 * 2);

  // Zero initial states + padded tail of enc history (for K=416 attention GEMMs)
  hipMemsetAsync(zero_h, 0, HB * 2, stream);
  hipMemsetAsync(c0, 0, HB * 4, stream);
  hipMemsetAsync(c1, 0, HB * 4, stream);
  hipMemsetAsync(enc_hist + 400LL * HB, 0, 16LL * HB * 2, stream);

  // Weight precast
  cast_bf16_k<<<(4096 * 512 / 4 + 255) / 256, 256, 0, stream>>>(eWih0, bWih0, 4096LL * 512);
  cast_bf16_k<<<(4096 * 1024 / 4 + 255) / 256, 256, 0, stream>>>(eWhh0, bWhh0, 4096LL * 1024);
  cast_bf16_k<<<(4096 * 1024 / 4 + 255) / 256, 256, 0, stream>>>(eWih1, bWih1, 4096LL * 1024);
  cast_bf16_k<<<(4096 * 1024 / 4 + 255) / 256, 256, 0, stream>>>(eWhh1, bWhh1, 4096LL * 1024);

  // Embeddings
  embed_enc_k<<<(400 * 32 * 512 + 255) / 256, 256, 0, stream>>>(contents, emb_enc, enc_x);
  embed_dec_k<<<(512 * 512 + 255) / 256, 256, 0, stream>>>(titles, emb_dec, dec_x);

  // Encoder layer 0 then layer 1 (layer-sequential == step-wise here)
  for (int t = 0; t < 400; ++t) {
    const bf16_t* hp = t ? h0_hist + (long long)(t - 1) * HB : zero_h;
    enc_lstm_step<512><<<dim3(64), 64, 0, stream>>>(enc_x + (long long)t * 32 * 512, hp,
                                                    bWih0, bWhh0, ebih0, ebhh0, c0,
                                                    h0_hist + (long long)t * HB);
  }
  for (int t = 0; t < 400; ++t) {
    const bf16_t* hp = t ? enc_hist + (long long)(t - 1) * HB : zero_h;
    enc_lstm_step<1024><<<dim3(64), 64, 0, stream>>>(h0_hist + (long long)t * HB, hp,
                                                     bWih1, bWhh1, ebih1, ebhh1, c1,
                                                     enc_hist + (long long)t * HB);
  }
  const bf16_t* hT0 = h0_hist + 399LL * HB;
  const bf16_t* hT1 = enc_hist + 399LL * HB;

  // Decoder: Hg[l] = hT[l] @ dWhh^T + bih + bhh   (M=32, N=4096, K=1024)
  gemm_wmma<0, 4><<<dim3(16, 2, 1), 128, 0, stream>>>(hT0, 1024, 0, dWhh0, 1024, 0,
                                                      Hg0, nullptr, 4096, 0, 4096, 1024,
                                                      dbih0, dbhh0, nullptr, 0, 0);
  gemm_wmma<0, 4><<<dim3(16, 2, 1), 128, 0, stream>>>(hT1, 1024, 0, dWhh1, 1024, 0,
                                                      Hg1, nullptr, 4096, 0, 4096, 1024,
                                                      dbih1, dbhh1, nullptr, 0, 0);
  // Batched independent decoder steps (M = 512)
  dec_lstm_step<512><<<dim3(64, 8), 128, 0, stream>>>(dec_x, dWih0, Hg0, c0, dec_h0);
  dec_lstm_step<1024><<<dim3(64, 8), 128, 0, stream>>>(dec_h0, dWih1, Hg1, c1, dec_o);

  // Attention
  // energy = dec_out @ w_in^T : [512,1024]
  gemm_wmma<0, 4><<<dim3(4, 32, 1), 128, 0, stream>>>(dec_o, 1024, 0, w_in, 1024, 0,
                                                      nullptr, energy, 1024, 0, 1024, 1024,
                                                      nullptr, nullptr, nullptr, 0, 0);
  // scores[b] = energy[b] @ enc_out[b]^T : per-batch [16,416], K=1024
  gemm_wmma<1, 1><<<dim3(7, 1, 32), 128, 0, stream>>>(energy, 1024, 16 * 1024,
                                                      enc_hist, 32768, 1024,
                                                      scores, nullptr, 416, 16 * 416, 416,
                                                      1024, nullptr, nullptr, nullptr, 0, 0);
  softmax400<<<512, 32, 0, stream>>>(scores, alpha);
  // context[b] = alpha[b] @ enc_out[b] : [16,1024], K=416 (padded, zero tails)
  gemm_wmma<2, 1><<<dim3(16, 1, 32), 128, 0, stream>>>(alpha, 416, 16 * 416,
                                                       enc_hist, 32768, 1024,
                                                       nullptr, context, 1024, 16 * 1024,
                                                       1024, 416, nullptr, nullptr,
                                                       nullptr, 0, 0);
  // attn_h = tanh([context|dec_out] @ w_outW^T + w_outb)  (two K-segments)
  gemm_wmma<0, 4><<<dim3(4, 32, 1), 128, 0, stream>>>(context, 1024, 0, w_outW, 2048, 0,
                                                      tmp1, nullptr, 1024, 0, 1024, 1024,
                                                      nullptr, nullptr, nullptr, 0, 0);
  gemm_wmma<0, 4><<<dim3(4, 32, 1), 128, 0, stream>>>(dec_o, 1024, 0, w_outW + 1024, 2048,
                                                      0, nullptr, attn_h, 1024, 0, 1024,
                                                      1024, w_outb, nullptr, tmp1, 1024, 1);
  // out = attn_h @ outW^T + outb : [512, 32000] fp32 -> d_out
  gemm_wmma<0, 4><<<dim3(125, 32, 1), 128, 0, stream>>>(attn_h, 1024, 0, outW, 1024, 0,
                                                        (float*)d_out, nullptr, 32000, 0,
                                                        32000, 1024,
                                                        outb, nullptr, nullptr, 0, 0);
  (void)in_sizes; (void)n_in; (void)out_size; (void)ws_size;
}
